// GHLALayer_62302795596608
// MI455X (gfx1250) — matrix operand order repaired and checked by reading, not compiled, not run on hardware
//
#include <hip/hip_runtime.h>
#include <hip/hip_bf16.h>

typedef __attribute__((ext_vector_type(16))) _Float16 v16h;
typedef __attribute__((ext_vector_type(8)))  float    v8f;
typedef __attribute__((ext_vector_type(4)))  unsigned int u32x4;
typedef __attribute__((ext_vector_type(8)))  int      i32x8;
typedef __attribute__((ext_vector_type(4)))  int      i32x4;

#define HD     16      // heads
#define TLEN   2048
#define BSZ    4
#define MROWS  (BSZ*TLEN)   // 8192
#define DMODEL 1024

// ---------------------------------------------------------------------------
// CDNA5 async / TDM helpers
// ---------------------------------------------------------------------------
__device__ __forceinline__ void async_g2l_b128(void* lds_dst, const void* gsrc) {
    uint32_t l = (uint32_t)(uintptr_t)lds_dst;   // low 32 bits of flat LDS addr = LDS offset
    uint64_t a = (uint64_t)(uintptr_t)gsrc;
    asm volatile("global_load_async_to_lds_b128 %0, %1, off"
                 :: "v"(l), "v"(a) : "memory");
}
__device__ __forceinline__ void wait_asynccnt0() {
    asm volatile("s_wait_asynccnt 0x0" ::: "memory");
}

// TDM: DMA a 128x32 f16 tile (row stride 1024 halves) into LDS with
// 16-DWORD interval / 4-DWORD padding => LDS row stride 40 halves (80 B).
__device__ __forceinline__ void tdm_load_A_tile(const _Float16* A, int m0, int k0,
                                                void* ldsbase) {
    uint64_t ga = (uint64_t)(uintptr_t)A + ((uint64_t)m0 * 1024u + (uint64_t)k0) * 2u;
    uint32_t la = (uint32_t)(uintptr_t)ldsbase;
    // D# group 0: count=1 | lds_addr | global_addr[56:0] | type=2
    u32x4 g0 = { 1u, la, (uint32_t)ga,
                 (uint32_t)((ga >> 32) & 0x01FFFFFFu) | (2u << 30) };
    // D# group 1: data_size=2B, pad_enable, pad_interval=16 DW, pad_amount=4 DW,
    // tensor_dim0=1024, tensor_dim1=8192, tile_dim0=32, tile_dim1=128,
    // tensor_dim0_stride=1024 (all in element units)
    i32x8 g1 = { (int)((1u << 16) | (1u << 20) | (3u << 22) | (3u << 25)),
                 (int)(1024u << 16),     // tensor_dim0[15:0] << 16 (bits 79:48)
                 (int)(8192u << 16),     // tensor_dim0 hi=0 | tensor_dim1[15:0]
                 (int)(32u << 16),       // tensor_dim1 hi=0 | tile_dim0=32
                 128,                    // tile_dim1=128 | tile_dim2=0
                 1024,                   // tensor_dim0_stride low
                 0, 0 };
    i32x4 gz4 = { 0, 0, 0, 0 };
    i32x8 gz8 = { 0, 0, 0, 0, 0, 0, 0, 0 };
    __builtin_amdgcn_tensor_load_to_lds(g0, g1, gz4, gz4, gz8, 0);
}

// ---------------------------------------------------------------------------
// f32 -> f16 cast (grid-stride)
// ---------------------------------------------------------------------------
__global__ void cvt_f32_f16(const float* __restrict__ src, _Float16* __restrict__ dst, int n) {
    int i = blockIdx.x * blockDim.x + threadIdx.x;
    int stride = gridDim.x * blockDim.x;
    for (; i < n; i += stride) dst[i] = (_Float16)src[i];
}

// ---------------------------------------------------------------------------
// WMMA GEMM: C(8192x1024) = A(8192x1024 f16 rm) * B(1024x1024 f16 rm)
// 256 thr = 8 waves; block tile 128x64; double-buffered LDS;
// A tile via TDM (TENSORcnt), B tile via regular staged transpose.
// Both fragment gathers are 16B-aligned contiguous -> ds_load_b128 pairs.
// mode 0: out[row*1024+col]; mode 1: q/k/v scatter; mode 2: +bias, sigmoid
// ---------------------------------------------------------------------------
__global__ __launch_bounds__(256)
void gemm_f16_wmma(const _Float16* __restrict__ A, const _Float16* __restrict__ Bm,
                   float* __restrict__ out, const float* __restrict__ bias, int mode) {
    __shared__ __align__(16) _Float16 As[2][128][40];  // 40 = 32 + 4-DWORD TDM pad
    __shared__ __align__(16) _Float16 Bs[2][64][40];   // transposed Bs[n][k], 80 B rows

    const int tid  = threadIdx.x;
    const int w    = tid >> 5;
    const int lane = tid & 31;
    const int lrow = lane & 15;
    const int m0   = blockIdx.x * 128;
    const int n0   = blockIdx.y * 64;

    v8f acc[4] = {};
    const uint32_t* B32 = (const uint32_t*)Bm;
    uint32_t breg[4];

    // ---- prologue: stage k0 = 0 into buffer 0 ----
    if (tid < 32) tdm_load_A_tile(A, m0, 0, &As[0][0][0]);
    #pragma unroll
    for (int s = 0; s < 4; ++s) {
        int idx = tid + s * 256;
        int k = idx >> 5, c = idx & 31;
        breg[s] = B32[k * 512 + (n0 >> 1) + c];
    }
    #pragma unroll
    for (int s = 0; s < 4; ++s) {
        int idx = tid + s * 256;
        int k = idx >> 5, c = idx & 31;
        Bs[0][2 * c][k]     = ((const _Float16*)&breg[s])[0];
        Bs[0][2 * c + 1][k] = ((const _Float16*)&breg[s])[1];
    }

    for (int k0 = 0; k0 < 1024; k0 += 32) {
        const int cur = (k0 >> 5) & 1;
        __builtin_amdgcn_s_wait_tensorcnt(0);
        __syncthreads();

        const bool more = (k0 + 32) < 1024;
        if (more) {
            if (tid < 32) tdm_load_A_tile(A, m0, k0 + 32, &As[cur ^ 1][0][0]);
            #pragma unroll
            for (int s = 0; s < 4; ++s) {
                int idx = tid + s * 256;
                int k = idx >> 5, c = idx & 31;
                breg[s] = B32[(k0 + 32 + k) * 512 + (n0 >> 1) + c];
            }
        }

        // A fragment (16-bit A 16x32, contiguous k order shared with B):
        // lanes 0-15: M=lane, K 0..15; lanes 16-31: M=lane-16, K 16..31
        union { v16h v; uint32_t u[8]; } af;
        {
            int asrow = w * 16 + lrow;
            int kb = (lane < 16) ? 0 : 16;
            #pragma unroll
            for (int j2 = 0; j2 < 4; ++j2)
                af.u[j2] = *(const uint32_t*)&As[cur][asrow][kb + 2 * j2];
            #pragma unroll
            for (int j2 = 0; j2 < 4; ++j2)
                af.u[4 + j2] = *(const uint32_t*)&As[cur][asrow][kb + 8 + 2 * j2];
        }
        // B fragment: lane gives N; lanes 0-15 K=0..15, lanes 16-31 K=16..31
        // 16 contiguous halves at 16B-aligned offsets -> two ds_load_b128
        #pragma unroll
        for (int nt = 0; nt < 4; ++nt) {
            union { v16h v; uint32_t u[8]; } bf;
            int n = nt * 16 + lrow;
            int kb2 = (lane < 16) ? 0 : 16;
            #pragma unroll
            for (int j2 = 0; j2 < 8; ++j2)
                bf.u[j2] = *(const uint32_t*)&Bs[cur][n][kb2 + 2 * j2];
            acc[nt] = __builtin_amdgcn_wmma_f32_16x16x32_f16(
                false, af.v, false, bf.v, (short)0, acc[nt], false, false);
        }

        if (more) {
            #pragma unroll
            for (int s = 0; s < 4; ++s) {
                int idx = tid + s * 256;
                int k = idx >> 5, c = idx & 31;
                Bs[cur ^ 1][2 * c][k]     = ((const _Float16*)&breg[s])[0];
                Bs[cur ^ 1][2 * c + 1][k] = ((const _Float16*)&breg[s])[1];
            }
        }
        __syncthreads();
    }

    // epilogue: C/D layout — lane 0-15: M=r, N=lane; lane 16-31: M=r+8, N=lane-16
    #pragma unroll
    for (int nt = 0; nt < 4; ++nt) {
        #pragma unroll
        for (int r = 0; r < 8; ++r) {
            int Mi  = (lane < 16) ? r : (r + 8);
            int row = m0 + w * 16 + Mi;
            int col = n0 + nt * 16 + lrow;
            float v = acc[nt][r];
            if (mode == 0) {
                out[row * 1024 + col] = v;
            } else {
                if (mode == 2)
                    v = __builtin_amdgcn_rcpf(1.0f + __expf(-(v + bias[col])));
                int b = row >> 11;          // / TLEN
                int t = row & 2047;
                int h = col >> 6;
                int d = col & 63;
                out[(((size_t)(b * HD + h) * TLEN) + t) * 64 + d] = v;
            }
        }
    }
}

// ---------------------------------------------------------------------------
// Sequential gated scan, one WG per (b,h), 256 threads.
// o_t = (q^T S_new) C_new - q^T G_new  (associativity kills the 64^3 matmul).
// Per-step q/k/v/aK/aC vectors double-buffered via async global->LDS loads,
// prefetching t+1 while computing t (hides HBM latency on the serial path).
// ---------------------------------------------------------------------------
__global__ __launch_bounds__(256)
void ghla_scan(const float* __restrict__ q_s, const float* __restrict__ k_s,
               const float* __restrict__ v_s, const float* __restrict__ aK_s,
               const float* __restrict__ aC_s, _Float16* __restrict__ o_pre) {
    __shared__ float S[64][66];
    __shared__ float Cq[64][66];
    __shared__ float G[64][66];
    __shared__ __align__(16) float vecs[2][5][64];   // q,k,v,aK,aC
    __shared__ float partA[4][64], partM[4][64], partG[4][64], partO[4][64];
    __shared__ float mvec[64];

    const int tid = threadIdx.x;
    const int bh  = blockIdx.x;        // 0..63
    const int b   = bh >> 4;
    const int h   = bh & 15;
    const int j   = tid & 63;
    const int g   = tid >> 6;
    const int i0  = g * 16;

    const size_t base = (size_t)bh * TLEN * 64;
    const float* qg  = q_s  + base;
    const float* kg  = k_s  + base;
    const float* vg  = v_s  + base;
    const float* aKg = aK_s + base;
    const float* aCg = aC_s + base;

    // prefetch t = 0 vectors: 5 arrays x 16 chunks of 16 B
    if (tid < 80) {
        int arr = tid >> 4, ch = tid & 15;
        const float* gp = (arr == 0) ? qg : (arr == 1) ? kg : (arr == 2) ? vg
                        : (arr == 3) ? aKg : aCg;
        async_g2l_b128(&vecs[0][arr][ch * 4], gp + ch * 4);
    }
    for (int idx = tid; idx < 64 * 66; idx += 256) {
        ((float*)S)[idx] = 0.f; ((float*)Cq)[idx] = 0.f; ((float*)G)[idx] = 0.f;
    }

    for (int t = 0; t < TLEN; ++t) {
        const int cur = t & 1;
        wait_asynccnt0();
        __syncthreads();
        if ((t + 1 < TLEN) && tid < 80) {
            int arr = tid >> 4, ch = tid & 15;
            const float* gp = (arr == 0) ? qg : (arr == 1) ? kg : (arr == 2) ? vg
                            : (arr == 3) ? aKg : aCg;
            async_g2l_b128(&vecs[cur ^ 1][arr][ch * 4], gp + (t + 1) * 64 + ch * 4);
        }
        const float* qv  = vecs[cur][0];
        const float* kv  = vecs[cur][1];
        const float* vv  = vecs[cur][2];
        const float* aKv = vecs[cur][3];
        const float* aCv = vecs[cur][4];

        // pass 1: C_prev = aC∘C, ktC partial, C_new = C_prev + q v^T, S_new
        float kj = kv[j], aKj = aKv[j], vj = vv[j];
        float p = 0.f;
        #pragma unroll
        for (int ii = 0; ii < 16; ++ii) {
            int i = i0 + ii;
            float cprev = aCv[i] * Cq[i][j];
            p += kv[i] * cprev;
            Cq[i][j] = cprev + qv[i] * vj;
            S[i][j]  = aKv[i] * aKj * S[i][j] + kv[i] * kj;
        }
        partA[g][j] = p;
        __syncthreads();

        // pass 2: G_new = aK∘G + k⊗ktC ; partials of q^T S_new and q^T G_new
        float ktC = partA[0][j] + partA[1][j] + partA[2][j] + partA[3][j];
        float pm = 0.f, pg = 0.f;
        #pragma unroll
        for (int ii = 0; ii < 16; ++ii) {
            int i = i0 + ii;
            float Gij = aKv[i] * G[i][j] + kv[i] * ktC;
            G[i][j] = Gij;
            pm += qv[i] * S[i][j];
            pg += qv[i] * Gij;
        }
        partM[g][j] = pm;
        partG[g][j] = pg;
        __syncthreads();

        if (g == 0) mvec[j] = partM[0][j] + partM[1][j] + partM[2][j] + partM[3][j];
        __syncthreads();

        // pass 3: o_v = m^T C_new - q^T G_new
        float po = 0.f;
        #pragma unroll
        for (int ii = 0; ii < 16; ++ii) {
            int r = i0 + ii;
            po += mvec[r] * Cq[r][j];
        }
        partO[g][j] = po;
        __syncthreads();

        if (g == 0) {
            float o = partO[0][j] + partO[1][j] + partO[2][j] + partO[3][j]
                    - (partG[0][j] + partG[1][j] + partG[2][j] + partG[3][j]);
            o_pre[(((size_t)b * TLEN + t) * HD + h) * 64 + j] = (_Float16)o;
        }
    }
}

// ---------------------------------------------------------------------------
extern "C" void kernel_launch(void* const* d_in, const int* in_sizes, int n_in,
                              void* d_out, int out_size, void* d_ws, size_t ws_size,
                              hipStream_t stream) {
    const float* x   = (const float*)d_in[0];
    const float* W_q = (const float*)d_in[1];
    const float* W_k = (const float*)d_in[2];
    const float* W_v = (const float*)d_in[3];
    const float* WgK = (const float*)d_in[4];
    const float* bgK = (const float*)d_in[5];
    const float* WgC = (const float*)d_in[6];
    const float* bgC = (const float*)d_in[7];
    const float* W_o = (const float*)d_in[8];
    float* out = (float*)d_out;

    char* ws = (char*)d_ws;
    size_t off = 0;
    auto take = [&](size_t bytes) -> char* {
        char* p = ws + off;
        off = (off + bytes + 255) & ~(size_t)255;
        return p;
    };

    const size_t NACT = (size_t)MROWS * DMODEL;   // 8192*1024
    const size_t NW   = (size_t)DMODEL * DMODEL;  // 1024*1024

    _Float16* x_h = (_Float16*)take(NACT * 2);
    _Float16* Wh[6];
    for (int i = 0; i < 6; ++i) Wh[i] = (_Float16*)take(NW * 2);
    float* q_s  = (float*)take(NACT * 4);
    float* k_s  = (float*)take(NACT * 4);
    float* v_s  = (float*)take(NACT * 4);
    float* aK_s = (float*)take(NACT * 4);
    float* aC_s = (float*)take(NACT * 4);
    _Float16* o_pre = (_Float16*)take(NACT * 2);

    // casts
    cvt_f32_f16<<<512, 256, 0, stream>>>(x, x_h, (int)NACT);
    cvt_f32_f16<<<256, 256, 0, stream>>>(W_q, Wh[0], (int)NW);
    cvt_f32_f16<<<256, 256, 0, stream>>>(W_k, Wh[1], (int)NW);
    cvt_f32_f16<<<256, 256, 0, stream>>>(W_v, Wh[2], (int)NW);
    cvt_f32_f16<<<256, 256, 0, stream>>>(WgK, Wh[3], (int)NW);
    cvt_f32_f16<<<256, 256, 0, stream>>>(WgC, Wh[4], (int)NW);
    cvt_f32_f16<<<256, 256, 0, stream>>>(W_o, Wh[5], (int)NW);

    // projections (WMMA + TDM-staged A tiles)
    dim3 gg(MROWS / 128, DMODEL / 64);
    gemm_f16_wmma<<<gg, 256, 0, stream>>>(x_h, Wh[0], q_s,  nullptr, 1);
    gemm_f16_wmma<<<gg, 256, 0, stream>>>(x_h, Wh[1], k_s,  nullptr, 1);
    gemm_f16_wmma<<<gg, 256, 0, stream>>>(x_h, Wh[2], v_s,  nullptr, 1);
    gemm_f16_wmma<<<gg, 256, 0, stream>>>(x_h, Wh[3], aK_s, bgK,     2);
    gemm_f16_wmma<<<gg, 256, 0, stream>>>(x_h, Wh[4], aC_s, bgC,     2);

    // sequential gated scan (async-prefetched inputs)
    ghla_scan<<<BSZ * HD, 256, 0, stream>>>(q_s, k_s, v_s, aK_s, aC_s, o_pre);

    // output projection (WMMA) -> d_out f32
    gemm_f16_wmma<<<gg, 256, 0, stream>>>(o_pre, Wh[5], out, nullptr, 0);
}
